// LigColumnAttention_7215545057293
// MI455X (gfx1250) — compile-verified, matrix operand order checked by reading, not run on hardware
//
#include <hip/hip_runtime.h>

typedef __attribute__((ext_vector_type(16))) _Float16 v16h;
typedef __attribute__((ext_vector_type(8)))  _Float16 v8h;
typedef __attribute__((ext_vector_type(4)))  _Float16 v4h;
typedef __attribute__((ext_vector_type(8)))  float    v8f;

// Problem dims
constexpr int SDIM  = 256;   // sequence (attended axis after transpose)
constexpr int RDIM  = 384;   // rows (batch-like axis)
constexpr int CDIM  = 256;   // channels
constexpr int NHEAD = 8;
constexpr int ACDIM = 32;    // per-head channels
constexpr int SHALF = 128;   // queries per workgroup

// LDS layout (f16 row strides chosen for 16B alignment + conflict-free frag loads)
constexpr int XN_RS = 264;   // 528B rows: 132 dw ≡ 4 mod 64 banks -> conflict free
constexpr int KH_RS = 40;    // 80B rows: 20 dw -> conflict free
constexpr int QH_RS = 40;
constexpr int VT_RS = 264;
constexpr int PS_RS = 40;

constexpr int XN_OFF = 0;
constexpr int XN_BY  = SDIM * XN_RS * 2;        // 135168
constexpr int OA_OFF = XN_OFF + XN_BY;          // fp32 out accumulator [128][256]
constexpr int OA_BY  = SHALF * CDIM * 4;        // 131072
constexpr int KH_OFF = OA_OFF + OA_BY;
constexpr int KH_BY  = SDIM * KH_RS * 2;        // 20480
constexpr int QH_OFF = KH_OFF + KH_BY;
constexpr int QH_BY  = SHALF * QH_RS * 2;       // 10240
constexpr int VT_OFF = QH_OFF + QH_BY;
constexpr int VT_BY  = ACDIM * VT_RS * 2;       // 16896
constexpr int PS_OFF = VT_OFF + VT_BY;
constexpr int PS_BY  = 8 * 16 * PS_RS * 2;      // 10240
constexpr int LDS_TOTAL = PS_OFF + PS_BY;       // 324096 <= 327680 (320KB/WGP)

__device__ __forceinline__ v8f wmma_f16(v16h a, v16h b, v8f c) {
    return __builtin_amdgcn_wmma_f32_16x16x32_f16(false, a, false, b, (short)0, c,
                                                  false, false);
}

// Load a 16x32 f16 A-style fragment (also B via B^T) from row-major f16 buffer.
// ISA 7.12.2 16-bit A layout: lanes 0-15 hold K0-7 / K16-23; lanes 16-31 hold
// K8-15 / K24-31 -> two contiguous 16B loads per lane.
__device__ __forceinline__ v16h load_frag(const _Float16* __restrict__ p,
                                          int row_stride, int row0, int col0) {
    const int lane = threadIdx.x & 31;
    const _Float16* q = p + (row0 + (lane & 15)) * row_stride + col0
                          + ((lane & 16) ? 8 : 0);
    v8h lo = *(const v8h*)q;
    v8h hi = *(const v8h*)(q + 16);
    return __builtin_shufflevector(lo, hi, 0,1,2,3,4,5,6,7,8,9,10,11,12,13,14,15);
}

// Reductions across a 16-lane half (keeps rows of the two C-layout halves apart)
__device__ __forceinline__ float rmax16(float v) {
    v = fmaxf(v, __shfl_xor(v, 1, 32));
    v = fmaxf(v, __shfl_xor(v, 2, 32));
    v = fmaxf(v, __shfl_xor(v, 4, 32));
    v = fmaxf(v, __shfl_xor(v, 8, 32));
    return v;
}
__device__ __forceinline__ float rsum16(float v) {
    v += __shfl_xor(v, 1, 32);
    v += __shfl_xor(v, 2, 32);
    v += __shfl_xor(v, 4, 32);
    v += __shfl_xor(v, 8, 32);
    return v;
}
__device__ __forceinline__ float rsum32(float v) {
    v += __shfl_xor(v, 1, 32);
    v += __shfl_xor(v, 2, 32);
    v += __shfl_xor(v, 4, 32);
    v += __shfl_xor(v, 8, 32);
    v += __shfl_xor(v, 16, 32);
    return v;
}

// ---- Prep: fp32 [K][N] weights -> f16 transposed [N][K]; q-scale folded in ----
__global__ void prep_weights_kernel(const float* __restrict__ wq,
                                    const float* __restrict__ wk,
                                    const float* __restrict__ wv,
                                    const float* __restrict__ wg,
                                    const float* __restrict__ wo,
                                    _Float16* __restrict__ wt) {
    const int m = blockIdx.x >> 8;        // which matrix (0..4)
    const int n = blockIdx.x & 255;       // output row (= input column)
    const float* w = (m == 0) ? wq : (m == 1) ? wk : (m == 2) ? wv
                   : (m == 3) ? wg : wo;
    const float scale = (m == 0) ? 0.17677669529663687f : 1.0f;  // 1/sqrt(32)
    _Float16* dst = wt + m * 65536 + n * 256;
    for (int k = threadIdx.x; k < 256; k += blockDim.x)
        dst[k] = (_Float16)(w[k * 256 + n] * scale);
}

// ---- Fused LN + gated column attention + output projection ----
__global__ __launch_bounds__(256, 1)
void col_attn_kernel(const float* __restrict__ msa,   // [1,S,R,C]
                     const float* __restrict__ mask,  // [1,S,R]
                     const float* __restrict__ ln_g,
                     const float* __restrict__ ln_b,
                     const float* __restrict__ bg,
                     const float* __restrict__ bo,
                     const _Float16* __restrict__ wt, // 5 transposed f16 weights
                     float* __restrict__ out) {       // [1,S,R,C]
    extern __shared__ char smem[];
    _Float16* XN = (_Float16*)(smem + XN_OFF);  // layernormed x, f16 [S][XN_RS]
    float*    OA = (float*)   (smem + OA_OFF);  // output accum fp32 [128][256]
    _Float16* KH = (_Float16*)(smem + KH_OFF);  // K head  [S][KH_RS]
    _Float16* QH = (_Float16*)(smem + QH_OFF);  // Q half  [128][QH_RS] (pre-scaled)
    _Float16* VT = (_Float16*)(smem + VT_OFF);  // V^T     [AC][VT_RS]
    _Float16* PS = (_Float16*)(smem + PS_OFF);  // per-wave 16x32 relayout scratch

    const int r    = blockIdx.x;
    const int half = blockIdx.y;           // which 128 query rows
    const int tid  = threadIdx.x;
    const int lane = tid & 31;
    const int w    = tid >> 5;             // wave 0..7
    const int hl   = (lane >> 4);          // lane half
    const int ln16 = lane & 15;

    const _Float16* WqT = wt;
    const _Float16* WkT = wt + 1 * 65536;
    const _Float16* WvT = wt + 2 * 65536;
    const _Float16* WgT = wt + 3 * 65536;
    const _Float16* WoT = wt + 4 * 65536;

    // Phase 0a: init output accumulator with bo
    for (int i = tid; i < SHALF * CDIM; i += 256) OA[i] = bo[i & (CDIM - 1)];

    // Phase 0b: load x[:, r, :] (one wave per row), layernorm over C, f16 -> LDS
    for (int s = w; s < SDIM; s += 8) {
        const float* row = msa + ((size_t)s * RDIM + r) * CDIM;
        float4 a = *(const float4*)(row + 4 * lane);
        float4 b = *(const float4*)(row + 128 + 4 * lane);
        float sum = a.x + a.y + a.z + a.w + b.x + b.y + b.z + b.w;
        float sq  = a.x*a.x + a.y*a.y + a.z*a.z + a.w*a.w
                  + b.x*b.x + b.y*b.y + b.z*b.z + b.w*b.w;
        sum = rsum32(sum);
        sq  = rsum32(sq);
        const float mu  = sum * (1.0f / CDIM);
        const float var = sq * (1.0f / CDIM) - mu * mu;
        const float inv = rsqrtf(var + 1e-5f);
        float4 g0 = *(const float4*)(ln_g + 4 * lane);
        float4 g1 = *(const float4*)(ln_g + 128 + 4 * lane);
        float4 b0 = *(const float4*)(ln_b + 4 * lane);
        float4 b1 = *(const float4*)(ln_b + 128 + 4 * lane);
        v4h o0, o1;
        o0[0] = (_Float16)((a.x - mu) * inv * g0.x + b0.x);
        o0[1] = (_Float16)((a.y - mu) * inv * g0.y + b0.y);
        o0[2] = (_Float16)((a.z - mu) * inv * g0.z + b0.z);
        o0[3] = (_Float16)((a.w - mu) * inv * g0.w + b0.w);
        o1[0] = (_Float16)((b.x - mu) * inv * g1.x + b1.x);
        o1[1] = (_Float16)((b.y - mu) * inv * g1.y + b1.y);
        o1[2] = (_Float16)((b.z - mu) * inv * g1.z + b1.z);
        o1[3] = (_Float16)((b.w - mu) * inv * g1.w + b1.w);
        *(v4h*)&XN[s * XN_RS + 4 * lane]       = o0;
        *(v4h*)&XN[s * XN_RS + 128 + 4 * lane] = o1;
    }
    __syncthreads();

    const int sq0 = half * SHALF + w * 16;  // this wave's global query-row base

    for (int h = 0; h < NHEAD; ++h) {
        const _Float16* wqh = WqT + h * ACDIM * 256;
        const _Float16* wkh = WkT + h * ACDIM * 256;
        const _Float16* wvh = WvT + h * ACDIM * 256;
        const _Float16* wgh = WgT + h * ACDIM * 256;

        // ---- K projection: wave -> M-tiles {2w, 2w+1} x N-tiles {0,1} ----
        for (int mt = 2 * w; mt <= 2 * w + 1; ++mt)
            for (int nt = 0; nt < 2; ++nt) {
                v8f acc = {};
                #pragma unroll
                for (int kc = 0; kc < 8; ++kc) {
                    v16h a = load_frag(XN, XN_RS, mt * 16, kc * 32);
                    v16h b = load_frag(wkh, 256, nt * 16, kc * 32);
                    acc = wmma_f16(a, b, acc);
                }
                const int rowb = mt * 16 + hl * 8, col = nt * 16 + ln16;
                #pragma unroll
                for (int j = 0; j < 8; ++j)
                    KH[(rowb + j) * KH_RS + col] = (_Float16)acc[j];
            }
        // ---- V projection, stored transposed [AC][S] ----
        for (int mt = 2 * w; mt <= 2 * w + 1; ++mt)
            for (int nt = 0; nt < 2; ++nt) {
                v8f acc = {};
                #pragma unroll
                for (int kc = 0; kc < 8; ++kc) {
                    v16h a = load_frag(XN, XN_RS, mt * 16, kc * 32);
                    v16h b = load_frag(wvh, 256, nt * 16, kc * 32);
                    acc = wmma_f16(a, b, acc);
                }
                const int colb = mt * 16 + hl * 8, rowv = nt * 16 + ln16;
                #pragma unroll
                for (int j = 0; j < 8; ++j)
                    VT[rowv * VT_RS + colb + j] = (_Float16)acc[j];
            }
        // ---- Q projection for this wave's slab (scale folded in WqT) ----
        for (int nt = 0; nt < 2; ++nt) {
            v8f acc = {};
            #pragma unroll
            for (int kc = 0; kc < 8; ++kc) {
                v16h a = load_frag(XN, XN_RS, sq0, kc * 32);
                v16h b = load_frag(wqh, 256, nt * 16, kc * 32);
                acc = wmma_f16(a, b, acc);
            }
            const int rowb = w * 16 + hl * 8, col = nt * 16 + ln16;
            #pragma unroll
            for (int j = 0; j < 8; ++j)
                QH[(rowb + j) * QH_RS + col] = (_Float16)acc[j];
        }
        __syncthreads();

        // ---- Flash attention over all 256 keys for the wave's 16 q rows ----
        v16h aQ = load_frag(QH, QH_RS, w * 16, 0);
        _Float16* myPS = PS + w * 16 * PS_RS;
        float m[8], l[8];
        v8f o0 = {}, o1 = {};
        #pragma unroll
        for (int j = 0; j < 8; ++j) { m[j] = -1e30f; l[j] = 0.0f; }

        for (int kc = 0; kc < 8; ++kc) {
            v16h bK0 = load_frag(KH, KH_RS, kc * 32, 0);
            v16h bK1 = load_frag(KH, KH_RS, kc * 32 + 16, 0);
            v8f z0 = {}, z1 = {};
            v8f s0 = wmma_f16(aQ, bK0, z0);
            v8f s1 = wmma_f16(aQ, bK1, z1);
            const float bias0 = 1e9f * (mask[(size_t)(kc * 32 + ln16) * RDIM + r] - 1.0f);
            const float bias1 = 1e9f * (mask[(size_t)(kc * 32 + 16 + ln16) * RDIM + r] - 1.0f);
            #pragma unroll
            for (int j = 0; j < 8; ++j) {
                float t0 = s0[j] + bias0, t1 = s1[j] + bias1;
                float mx = rmax16(fmaxf(t0, t1));
                float mn = fmaxf(m[j], mx);
                float sc = __expf(m[j] - mn);
                m[j] = mn;
                float p0 = __expf(t0 - mn), p1 = __expf(t1 - mn);
                l[j] = l[j] * sc + rsum16(p0 + p1);
                o0[j] *= sc; o1[j] *= sc;
                myPS[(hl * 8 + j) * PS_RS + ln16]      = (_Float16)p0;
                myPS[(hl * 8 + j) * PS_RS + 16 + ln16] = (_Float16)p1;
            }
            v16h aP  = load_frag(myPS, PS_RS, 0, 0);          // same-wave, DS in-order
            v16h bV0 = load_frag(VT, VT_RS, 0,  kc * 32);
            v16h bV1 = load_frag(VT, VT_RS, 16, kc * 32);
            o0 = wmma_f16(aP, bV0, o0);
            o1 = wmma_f16(aP, bV1, o1);
        }

        // ---- Gate = sigmoid(XN_slab @ Wg_h + bg), fused with 1/l normalize ----
        v8f g0 = {}, g1 = {};
        #pragma unroll
        for (int kc = 0; kc < 8; ++kc) {
            v16h aX = load_frag(XN, XN_RS, sq0, kc * 32);
            g0 = wmma_f16(aX, load_frag(wgh, 256, 0,  kc * 32), g0);
            g1 = wmma_f16(aX, load_frag(wgh, 256, 16, kc * 32), g1);
        }
        const float bg0 = bg[h * ACDIM + ln16];
        const float bg1 = bg[h * ACDIM + 16 + ln16];
        #pragma unroll
        for (int j = 0; j < 8; ++j) {
            float invl = 1.0f / l[j];
            float gg0 = 1.0f / (1.0f + __expf(-(g0[j] + bg0)));
            float gg1 = 1.0f / (1.0f + __expf(-(g1[j] + bg1)));
            float r0 = o0[j] * invl * gg0;
            float r1 = o1[j] * invl * gg1;
            myPS[(hl * 8 + j) * PS_RS + ln16]      = (_Float16)r0;
            myPS[(hl * 8 + j) * PS_RS + 16 + ln16] = (_Float16)r1;
        }

        // ---- Output projection: OA[slab, :] += gatedO(16x32) @ Wo_h(32x256) ----
        v16h aO = load_frag(myPS, PS_RS, 0, 0);
        for (int nt = 0; nt < 16; ++nt) {
            v16h bO = load_frag(WoT, 256, nt * 16, h * ACDIM);
            v8f z = {};
            v8f acc = wmma_f16(aO, bO, z);
            const int rowb = w * 16 + hl * 8, col = nt * 16 + ln16;
            #pragma unroll
            for (int j = 0; j < 8; ++j)
                atomicAdd(&OA[(rowb + j) * CDIM + col], acc[j]);
        }
        __syncthreads();   // protect KH/QH/VT before next head overwrites
    }

    // ---- Write out: out[0, s, r, :] = OA[s - half*128, :] ----
    for (int sl = w; sl < SHALF; sl += 8) {
        const int s = half * SHALF + sl;
        float4 a = *(const float4*)&OA[sl * CDIM + 4 * lane];
        float4 b = *(const float4*)&OA[sl * CDIM + 128 + 4 * lane];
        float* dst = out + ((size_t)s * RDIM + r) * CDIM;
        *(float4*)(dst + 4 * lane)       = a;
        *(float4*)(dst + 128 + 4 * lane) = b;
    }
}

extern "C" void kernel_launch(void* const* d_in, const int* in_sizes, int n_in,
                              void* d_out, int out_size, void* d_ws, size_t ws_size,
                              hipStream_t stream) {
    const float* msa  = (const float*)d_in[0];
    const float* mask = (const float*)d_in[1];
    const float* ln_g = (const float*)d_in[2];
    const float* ln_b = (const float*)d_in[3];
    const float* wq   = (const float*)d_in[4];
    const float* wk   = (const float*)d_in[5];
    const float* wv   = (const float*)d_in[6];
    const float* wg   = (const float*)d_in[7];
    const float* bg   = (const float*)d_in[8];
    const float* wo   = (const float*)d_in[9];
    const float* bo   = (const float*)d_in[10];
    _Float16* wt = (_Float16*)d_ws;   // 5 * 65536 f16 = 640KB scratch

    prep_weights_kernel<<<dim3(5 * 256), 256, 0, stream>>>(wq, wk, wv, wg, wo, wt);
    col_attn_kernel<<<dim3(RDIM, 2), 256, LDS_TOTAL, stream>>>(
        msa, mask, ln_g, ln_b, bg, bo, wt, (float*)d_out);
}
